// MultiHeadAttention_27384711479714
// MI455X (gfx1250) — matrix operand order repaired and checked by reading, not compiled, not run on hardware
//
#include <hip/hip_runtime.h>

typedef __bf16 bf16_t;
typedef __attribute__((ext_vector_type(16))) __bf16 v16bf;
typedef __attribute__((ext_vector_type(8)))  __bf16 v8bf;
typedef __attribute__((ext_vector_type(8)))  float  v8f;
typedef __attribute__((ext_vector_type(4)))  int    v4i;

#define BATCH   2
#define TSEQ    2048
#define DMODEL  1024
#define NHEAD   16
#define HD      64

// GEMM tiling: 128x128 block tile, K-chunk 32, 8 waves (2x4), double-buffered
#define BM 128
#define BN 128
#define BK 32
#define LDT 40           // LDS row stride (bf16): 80 B, 16-B aligned
#define NKC (DMODEL / BK)

// attention tiling: 128 Q rows per block (16 per wave), 64-wide K/V chunks
#define AK 64
#define LDK 72           // 144 B row stride, 16-B aligned
#define NAC (TSEQ / AK)

// ---------------------------------------------------------------------------
// CDNA5 async global->LDS copies (ASYNCcnt), with sync fallback so the file
// always compiles.  Each call moves 16 bytes per lane, memory -> LDS directly.
// Builtin signature (from hipcc diagnostic): (global v4i*, lds v4i*, imm, imm).
// ---------------------------------------------------------------------------
#if __has_builtin(__builtin_amdgcn_global_load_async_to_lds_b128) && \
    __has_builtin(__builtin_amdgcn_s_wait_asynccnt)
#define USE_ASYNC 1
#else
#define USE_ASYNC 0
#endif

static __device__ __forceinline__ void async_copy16(const bf16_t* g, bf16_t* l) {
#if USE_ASYNC
  __builtin_amdgcn_global_load_async_to_lds_b128(
      (__attribute__((address_space(1))) v4i*)g,
      (__attribute__((address_space(3))) v4i*)l, 0, 0);
#else
  *(v8bf*)l = *(const v8bf*)g;
#endif
}
static __device__ __forceinline__ void wait_async_keep4() {
#if USE_ASYNC
  __builtin_amdgcn_s_wait_asynccnt(4);   // allow next tile's 4 copies in flight
#endif
}
static __device__ __forceinline__ void wait_async_all() {
#if USE_ASYNC
  __builtin_amdgcn_s_wait_asynccnt(0);
#endif
}

// ---------------------------------------------------------------------------
// WMMA bf16 fragment loaders (16x16x32), ISA 7.12.2 layouts.
// A: lanes 0-15 row M=lane, K={0..7,16..23}; lanes 16-31 K={8..15,24..31}.
// B: lanes 0-15 col N=lane, K={0..7,16..23}; lanes 16-31 K={8..15,24..31}.
// ---------------------------------------------------------------------------
static __device__ __forceinline__ v16bf frag_a(const bf16_t* tile, int row0,
                                               int koff, int ld, int lane) {
  const bf16_t* p = tile + (size_t)(row0 + (lane & 15)) * ld + koff + ((lane >> 4) << 3);
  v8bf lo = *(const v8bf*)(p);
  v8bf hi = *(const v8bf*)(p + 16);
  v16bf a;
#pragma unroll
  for (int i = 0; i < 8; ++i) { a[i] = lo[i]; a[8 + i] = hi[i]; }
  return a;
}

static __device__ __forceinline__ v16bf frag_b(const bf16_t* tile, int col0,
                                               int koff, int ld, int lane) {
  const bf16_t* p = tile + (size_t)(col0 + (lane & 15)) * ld + koff + ((lane >> 4) << 3);
  v8bf lo = *(const v8bf*)(p);
  v8bf hi = *(const v8bf*)(p + 16);
  v16bf b;
#pragma unroll
  for (int i = 0; i < 8; ++i) { b[i] = lo[i]; b[8 + i] = hi[i]; }
  return b;
}

// ---------------------------------------------------------------------------
// One-time fp32 -> bf16 row-major pack (8 elems / thread).
// ---------------------------------------------------------------------------
__global__ __launch_bounds__(256)
void to_bf16(const float* __restrict__ src, bf16_t* __restrict__ dst)
{
  size_t i = (size_t)blockIdx.x * 256 + threadIdx.x;   // 8-elem chunk index
  const float4* s = (const float4*)src + i * 2;
  float4 f0 = s[0], f1 = s[1];
  v8bf h;
  h[0]=(bf16_t)f0.x; h[1]=(bf16_t)f0.y; h[2]=(bf16_t)f0.z; h[3]=(bf16_t)f0.w;
  h[4]=(bf16_t)f1.x; h[5]=(bf16_t)f1.y; h[6]=(bf16_t)f1.z; h[7]=(bf16_t)f1.w;
  *(v8bf*)(dst + i * 8) = h;
}

// ---------------------------------------------------------------------------
// One-time fp32 W[K][N] -> bf16 Wt[N][K], 64x64 tiles through LDS so both
// the global read and the global write are coalesced.
// ---------------------------------------------------------------------------
__global__ __launch_bounds__(256)
void w_transpose(const float* __restrict__ W, bf16_t* __restrict__ Wt)
{
  __shared__ bf16_t tile[64][LDK];
  const int tid = threadIdx.x;
  const int n0 = blockIdx.x * 64;
  const int k0 = blockIdx.y * 64;
#pragma unroll
  for (int i = 0; i < 4; ++i) {
    int idx = tid + i * 256;          // 0..1023 float4 chunks
    int r   = idx >> 4;               // 0..63 (k)
    int c4  = (idx & 15) << 2;        // 0..60 (n)
    float4 f = *(const float4*)(W + (size_t)(k0 + r) * DMODEL + n0 + c4);
    tile[c4 + 0][r] = (bf16_t)f.x;
    tile[c4 + 1][r] = (bf16_t)f.y;
    tile[c4 + 2][r] = (bf16_t)f.z;
    tile[c4 + 3][r] = (bf16_t)f.w;
  }
  __syncthreads();
#pragma unroll
  for (int i = 0; i < 2; ++i) {
    int idx = tid + i * 256;          // 0..511 chunks of 8
    int r   = idx >> 3;               // 0..63 (n)
    int c8  = (idx & 7) << 3;         // 0..56 (k)
    *(v8bf*)(Wt + (size_t)(n0 + r) * DMODEL + k0 + c8) = *(const v8bf*)(&tile[r][c8]);
  }
}

// ---------------------------------------------------------------------------
// Double-buffered async GEMM: Out = (A[M,K]*Wt[N,K]^T + bias) * sc
// A, Wt are bf16 in global memory; mainloop is pure async-DMA + WMMA.
// OUT_MODE: 0 = bf16 heads layout [b,h,t,d] (Q,K)
//           1 = bf16 transposed heads [b,h,d,t] (V)
//           2 = fp32 row-major (final projection)
// ---------------------------------------------------------------------------
template <int OUT_MODE>
__global__ __launch_bounds__(256)
void mha_gemm(const bf16_t* __restrict__ A, const bf16_t* __restrict__ Wt,
              const float* __restrict__ bias, void* __restrict__ Out, float sc)
{
  __shared__ bf16_t sA[2][BM * LDT];
  __shared__ bf16_t sW[2][BN * LDT];

  const int tid  = threadIdx.x;
  const int lane = tid & 31;
  const int wave = tid >> 5;
  const int wm   = wave >> 2;       // 0..1 -> 64 rows
  const int wn   = wave & 3;        // 0..3 -> 32 cols
  const int bm0  = blockIdx.y * BM;
  const int bn0  = blockIdx.x * BN;

  const v8f vz = {};
  v8f acc[4][2];
#pragma unroll
  for (int i = 0; i < 4; ++i)
#pragma unroll
    for (int j = 0; j < 2; ++j) acc[i][j] = vz;

  auto stage = [&](int buf, int k0) {           // 4 async b128 per thread
#pragma unroll
    for (int i = 0; i < 2; ++i) {
      int idx = tid + i * 256;                  // 0..511
      int r   = idx >> 2;                       // 0..127
      int c8  = (idx & 3) << 3;                 // 0,8,16,24
      async_copy16(A  + (size_t)(bm0 + r) * DMODEL + k0 + c8, &sA[buf][r * LDT + c8]);
      async_copy16(Wt + (size_t)(bn0 + r) * DMODEL + k0 + c8, &sW[buf][r * LDT + c8]);
    }
  };

  auto compute = [&](int buf) {
    v16bf af[4];
#pragma unroll
    for (int mt = 0; mt < 4; ++mt) af[mt] = frag_a(sA[buf], wm * 64 + mt * 16, 0, LDT, lane);
    v16bf bfr[2];
#pragma unroll
    for (int nt = 0; nt < 2; ++nt) bfr[nt] = frag_b(sW[buf], wn * 32 + nt * 16, 0, LDT, lane);
#pragma unroll
    for (int mt = 0; mt < 4; ++mt)
#pragma unroll
      for (int nt = 0; nt < 2; ++nt)
        acc[mt][nt] = __builtin_amdgcn_wmma_f32_16x16x32_bf16(
            false, af[mt], false, bfr[nt], (short)0, acc[mt][nt], false, false);
  };

  stage(0, 0);
  for (int kc = 0; kc < NKC - 1; ++kc) {
    int cur = kc & 1;
    stage(cur ^ 1, (kc + 1) * BK);   // prefetch next tile while computing
    wait_async_keep4();              // current tile's copies done
    __syncthreads();
    compute(cur);
    __syncthreads();                 // everyone done reading before overwrite
  }
  wait_async_all();
  __syncthreads();
  compute((NKC - 1) & 1);

  // ---- epilogue -----------------------------------------------------------
  const int nl = lane & 15;
  const int kh = lane >> 4;
#pragma unroll
  for (int mt = 0; mt < 4; ++mt) {
#pragma unroll
    for (int nt = 0; nt < 2; ++nt) {
      int gn = bn0 + wn * 32 + nt * 16 + nl;
      float bv = bias[gn];
#pragma unroll
      for (int r = 0; r < 8; ++r) {
        int gm = bm0 + wm * 64 + mt * 16 + r + (kh << 3);
        float val = (acc[mt][nt][r] + bv) * sc;
        if (OUT_MODE == 2) {
          ((float*)Out)[(size_t)gm * DMODEL + gn] = val;
        } else {
          int b = gm >> 11, t = gm & (TSEQ - 1);
          int h = gn >> 6,  d = gn & (HD - 1);
          size_t bh = (size_t)b * NHEAD + h;
          if (OUT_MODE == 0)
            ((bf16_t*)Out)[((bh * TSEQ + t) << 6) + d] = (bf16_t)val;        // [bh][t][d]
          else
            ((bf16_t*)Out)[(bh * HD + d) * TSEQ + t] = (bf16_t)val;          // [bh][d][t]
        }
      }
    }
  }
}

// ---------------------------------------------------------------------------
// Flash attention, double-buffered async K/V staging.
// One block = one (b,h) x 128 Q rows; one wave = 16 Q rows, Q resident.
// ---------------------------------------------------------------------------
__global__ __launch_bounds__(256)
void mha_attn(const bf16_t* __restrict__ Qb, const bf16_t* __restrict__ Kb,
              const bf16_t* __restrict__ Vt, bf16_t* __restrict__ Ob)
{
  __shared__ bf16_t sK [2][AK * LDK];   // [kpos][d]
  __shared__ bf16_t sVt[2][HD * LDK];   // [d][kpos] (Vt already transposed)
  __shared__ bf16_t sP [8][16 * LDK];   // per-wave P staging [row][kpos]

  const int tid  = threadIdx.x;
  const int lane = tid & 31;
  const int wave = tid >> 5;
  const int nl   = lane & 15;
  const int kh   = lane >> 4;
  const int bh   = blockIdx.y;
  const int b    = bh >> 4;
  const int h    = bh & 15;
  const int q0   = blockIdx.x * 128 + wave * 16;

  const bf16_t* Qp  = Qb + (size_t)bh * TSEQ * HD;
  const bf16_t* Kp  = Kb + (size_t)bh * TSEQ * HD;
  const bf16_t* Vtp = Vt + (size_t)bh * HD * TSEQ;

  // resident, pre-scaled Q fragments (two 16x32 A frags)
  v16bf qf[2];
  {
    const bf16_t* qrow = Qp + (size_t)(q0 + nl) * HD + (kh << 3);
#pragma unroll
    for (int c = 0; c < 2; ++c) {
      v8bf lo = *(const v8bf*)(qrow + c * 32);
      v8bf hi = *(const v8bf*)(qrow + c * 32 + 16);
#pragma unroll
      for (int i = 0; i < 8; ++i) { qf[c][i] = lo[i]; qf[c][8 + i] = hi[i]; }
    }
  }

  const v8f vz = {};
  float mrow[8], lrow[8];
  v8f o[4];
#pragma unroll
  for (int r = 0; r < 8; ++r) { mrow[r] = -1e30f; lrow[r] = 0.f; }
#pragma unroll
  for (int nt = 0; nt < 4; ++nt) o[nt] = vz;

  auto stageKV = [&](int buf, int kt) {         // 4 async b128 per thread
#pragma unroll
    for (int i = 0; i < 2; ++i) {
      int idx = tid + i * 256;                  // 0..511
      int r   = idx >> 3;                       // 0..63
      int c8  = (idx & 7) << 3;                 // 0..56
      async_copy16(Kp  + (size_t)(kt + r) * HD   + c8, &sK [buf][r * LDK + c8]);
      async_copy16(Vtp + (size_t)r * TSEQ + kt   + c8, &sVt[buf][r * LDK + c8]);
    }
  };

  bf16_t* myP = sP[wave];

  auto chunk = [&](int buf) {
    // S = Q * K^T
    v8f s[4];
#pragma unroll
    for (int nt = 0; nt < 4; ++nt) s[nt] = vz;
#pragma unroll
    for (int c = 0; c < 2; ++c) {
#pragma unroll
      for (int nt = 0; nt < 4; ++nt) {
        v16bf kb = frag_b(sK[buf], nt * 16, c * 32, LDK, lane);
        s[nt] = __builtin_amdgcn_wmma_f32_16x16x32_bf16(
            false, qf[c], false, kb, (short)0, s[nt], false, false);
      }
    }
    // online softmax (scale already folded into Q)
#pragma unroll
    for (int r = 0; r < 8; ++r) {
      float t = fmaxf(fmaxf(s[0][r], s[1][r]), fmaxf(s[2][r], s[3][r]));
      t = fmaxf(t, __shfl_xor(t, 1, 32));
      t = fmaxf(t, __shfl_xor(t, 2, 32));
      t = fmaxf(t, __shfl_xor(t, 4, 32));
      t = fmaxf(t, __shfl_xor(t, 8, 32));
      float mn   = fmaxf(mrow[r], t);
      float corr = __expf(mrow[r] - mn);
      mrow[r] = mn;
      float rs = 0.f;
#pragma unroll
      for (int nt = 0; nt < 4; ++nt) { s[nt][r] = __expf(s[nt][r] - mn); rs += s[nt][r]; }
      rs += __shfl_xor(rs, 1, 32);
      rs += __shfl_xor(rs, 2, 32);
      rs += __shfl_xor(rs, 4, 32);
      rs += __shfl_xor(rs, 8, 32);
      lrow[r] = lrow[r] * corr + rs;
#pragma unroll
      for (int nt = 0; nt < 4; ++nt) o[nt][r] *= corr;
#pragma unroll
      for (int nt = 0; nt < 4; ++nt)
        myP[(r + (kh << 3)) * LDK + nt * 16 + nl] = (bf16_t)s[nt][r];
    }
    // O += P * V  (same-wave LDS RAW handled by DScnt waits)
#pragma unroll
    for (int c = 0; c < 2; ++c) {
      v16bf pa = frag_a(myP, 0, c * 32, LDK, lane);
#pragma unroll
      for (int nt = 0; nt < 4; ++nt) {
        v16bf vb = frag_b(sVt[buf], nt * 16, c * 32, LDK, lane);
        o[nt] = __builtin_amdgcn_wmma_f32_16x16x32_bf16(
            false, pa, false, vb, (short)0, o[nt], false, false);
      }
    }
  };

  stageKV(0, 0);
  for (int kc = 0; kc < NAC - 1; ++kc) {
    int cur = kc & 1;
    stageKV(cur ^ 1, (kc + 1) * AK);
    wait_async_keep4();
    __syncthreads();
    chunk(cur);
    __syncthreads();
  }
  wait_async_all();
  __syncthreads();
  chunk((NAC - 1) & 1);

  // normalize + store bf16 in [b][t][h*64+d]
#pragma unroll
  for (int r = 0; r < 8; ++r) {
    float inv = __builtin_amdgcn_rcpf(lrow[r]);
    int t = q0 + r + (kh << 3);
#pragma unroll
    for (int nt = 0; nt < 4; ++nt) {
      int d = nt * 16 + nl;
      Ob[(size_t)(b * TSEQ + t) * DMODEL + h * HD + d] = (bf16_t)(o[nt][r] * inv);
    }
  }
}

// ---------------------------------------------------------------------------
extern "C" void kernel_launch(void* const* d_in, const int* in_sizes, int n_in,
                              void* d_out, int out_size, void* d_ws, size_t ws_size,
                              hipStream_t stream)
{
  (void)in_sizes; (void)n_in; (void)out_size; (void)ws_size;
  const float* q  = (const float*)d_in[0];
  const float* k  = (const float*)d_in[1];
  const float* v  = (const float*)d_in[2];
  const float* Wq = (const float*)d_in[3];
  const float* bq = (const float*)d_in[4];
  const float* Wk = (const float*)d_in[5];
  const float* bk = (const float*)d_in[6];
  const float* Wv = (const float*)d_in[7];
  const float* bv = (const float*)d_in[8];
  const float* Wo = (const float*)d_in[9];
  const float* bo = (const float*)d_in[10];
  float* out = (float*)d_out;

  const size_t S1 = (size_t)BATCH * TSEQ * DMODEL;   // 4M elems
  const size_t S2 = (size_t)DMODEL * DMODEL;         // 1M elems
  bf16_t* qb  = (bf16_t*)d_ws;
  bf16_t* kb  = qb  + S1;
  bf16_t* vb  = kb  + S1;
  bf16_t* Wqt = vb  + S1;
  bf16_t* Wkt = Wqt + S2;
  bf16_t* Wvt = Wkt + S2;
  bf16_t* Wot = Wvt + S2;
  bf16_t* Qh  = Wot + S2;      // [b,h,t,d]
  bf16_t* Kh  = Qh  + S1;      // [b,h,t,d]
  bf16_t* Vth = Kh  + S1;      // [b,h,d,t]
  bf16_t* Ab  = Vth + S1;      // [b*t, 1024]

  dim3 blk(256, 1, 1);

  // one-time packs / transposes (bandwidth-trivial: ~60 MB total)
  dim3 gc(S1 / (256 * 8), 1, 1);
  to_bf16<<<gc, blk, 0, stream>>>(q, qb);
  to_bf16<<<gc, blk, 0, stream>>>(k, kb);
  to_bf16<<<gc, blk, 0, stream>>>(v, vb);
  dim3 gt(DMODEL / 64, DMODEL / 64, 1);
  w_transpose<<<gt, blk, 0, stream>>>(Wq, Wqt);
  w_transpose<<<gt, blk, 0, stream>>>(Wk, Wkt);
  w_transpose<<<gt, blk, 0, stream>>>(Wv, Wvt);
  w_transpose<<<gt, blk, 0, stream>>>(Wo, Wot);

  // projections (scale folded into Q), attention, output projection
  dim3 gg(DMODEL / BN, (BATCH * TSEQ) / BM, 1);
  mha_gemm<0><<<gg, blk, 0, stream>>>(qb, Wqt, bq, Qh,  0.125f);
  mha_gemm<0><<<gg, blk, 0, stream>>>(kb, Wkt, bk, Kh,  1.0f);
  mha_gemm<1><<<gg, blk, 0, stream>>>(vb, Wvt, bv, Vth, 1.0f);

  dim3 ga(TSEQ / 128, BATCH * NHEAD, 1);
  mha_attn<<<ga, blk, 0, stream>>>(Qh, Kh, Vth, Ab);

  mha_gemm<2><<<gg, blk, 0, stream>>>(Ab, Wot, bo, out, 1.0f);
}
